// SlidingWindowAttention_53042846106010
// MI455X (gfx1250) — compile-verified, matrix operand order checked by reading, not run on hardware
//
#include <hip/hip_runtime.h>
#include <stdint.h>

// ---------------------------------------------------------------------------
// Types / helpers
// ---------------------------------------------------------------------------
typedef unsigned short u16b;
typedef __attribute__((ext_vector_type(16))) __bf16 v16bf;
typedef __attribute__((ext_vector_type(8)))  float  v8f;
typedef int v4i_t __attribute__((vector_size(16)));

union FragB { v16bf v; uint4 u[2]; };

__device__ __forceinline__ u16b f2bf(float f) {
  unsigned u = __float_as_uint(f);
  unsigned r = (u + 0x7FFFu + ((u >> 16) & 1u)) >> 16;   // round-to-nearest-even
  return (u16b)r;
}

__device__ __forceinline__ v8f wmma_bf16(v16bf a, v16bf b, v8f c) {
  // D(f32,16x16) = A(bf16,16x32) * B(bf16,32x16) + C
  return __builtin_amdgcn_wmma_f32_16x16x32_bf16(false, a, false, b, (short)0, c,
                                                 false, false);
}

__device__ __forceinline__ v8f zero8() {
  v8f z;
#pragma unroll
  for (int e = 0; e < 8; ++e) z[e] = 0.0f;
  return z;
}

// lane-xor shuffle within 16-lane groups via ds_swizzle (imm must be literal)
#define SWZ_XOR_F(x, m) \
  __int_as_float(__builtin_amdgcn_ds_swizzle(__float_as_int(x), (0x1f | ((m) << 10))))

// ---------------------------------------------------------------------------
// CDNA5 async global->LDS copy (ASYNCcnt path), with synchronous fallback.
// ---------------------------------------------------------------------------
#if defined(__has_builtin)
#if __has_builtin(__builtin_amdgcn_global_load_async_to_lds_b128) && \
    __has_builtin(__builtin_amdgcn_s_wait_asynccnt)
#define USE_ASYNC_LDS 1
#endif
#endif
#ifndef USE_ASYNC_LDS
#define USE_ASYNC_LDS 0
#endif

__device__ __forceinline__ void async_cp16(void* lds_dst, const void* gsrc) {
#if USE_ASYNC_LDS
  __builtin_amdgcn_global_load_async_to_lds_b128(
      (__attribute__((address_space(1))) v4i_t*)gsrc,
      (__attribute__((address_space(3))) v4i_t*)lds_dst, 0, 0);
#else
  *(uint4*)lds_dst = *(const uint4*)gsrc;
#endif
}

__device__ __forceinline__ void async_wait0() {
#if USE_ASYNC_LDS
  __builtin_amdgcn_s_wait_asynccnt(0);
#endif
}

// ---------------------------------------------------------------------------
// Pre-pass: fp32 -> bf16 conversions
// ---------------------------------------------------------------------------
__global__ void cvt_f32_bf16_kernel(const float* __restrict__ in,
                                    u16b* __restrict__ out, int n) {
  int i = blockIdx.x * blockDim.x + threadIdx.x;
  int stride = gridDim.x * blockDim.x;
  for (; i < n; i += stride) out[i] = f2bf(in[i]);
}

// W[K][N] fp32 -> Wt[N][K] bf16
__global__ void transpose_cvt_kernel(const float* __restrict__ W,
                                     u16b* __restrict__ Wt, int K, int N) {
  int i = blockIdx.x * blockDim.x + threadIdx.x;
  int stride = gridDim.x * blockDim.x;
  int total = K * N;
  for (; i < total; i += stride) {
    int k = i / N, n = i - k * N;
    Wt[(size_t)n * K + k] = f2bf(W[i]);
  }
}

// ---------------------------------------------------------------------------
// bf16 GEMM: C[M][N] = A[M][K] * Bt[N][K]^T + bias
// Block tile 128x128, BK=32, 256 threads = 8 waves, wave tile 32x64.
// Double-buffered async global->LDS staging overlapped with WMMA.
// MODE 0: QKV epilogue -> scatter bf16 into q/k (row-major) and vt (transposed)
// MODE 1: fp32 output with bias (final projection)
// ---------------------------------------------------------------------------
template <int MODE>
__global__ __launch_bounds__(256) void gemm_bf16_kernel(
    const u16b* __restrict__ A, const u16b* __restrict__ Bt,
    const float* __restrict__ bias, u16b* __restrict__ qo,
    u16b* __restrict__ ko, u16b* __restrict__ vto, float* __restrict__ outf,
    int M, int N, int K) {
  __shared__ u16b As[2][128][40];  // [buf][row][k], pad 8 halfs
  __shared__ u16b Bs[2][128][40];  // Bt rows (= B columns)

  const int tid = threadIdx.x;
  const int lane = tid & 31, wv = tid >> 5;
  const int hl = lane >> 4, nn = lane & 15;
  const int wm = wv >> 1, wn = wv & 1;  // 4x2 wave grid
  const int m0 = blockIdx.y * 128, n0 = blockIdx.x * 128;

  v8f acc[2][4];
#pragma unroll
  for (int i = 0; i < 2; ++i)
#pragma unroll
    for (int j = 0; j < 4; ++j) acc[i][j] = zero8();

  // 512 chunks of 8 halfs per operand tile; 2 per thread
  const int row0 = (tid * 2) >> 2, c80 = ((tid * 2) & 3) << 3;
  const int row1 = (tid * 2 + 1) >> 2, c81 = ((tid * 2 + 1) & 3) << 3;

#define STAGE_TILE(buf, k0)                                                    \
  do {                                                                         \
    async_cp16(&As[buf][row0][c80], &A[(size_t)(m0 + row0) * K + (k0) + c80]); \
    async_cp16(&Bs[buf][row0][c80], &Bt[(size_t)(n0 + row0) * K + (k0) + c80]);\
    async_cp16(&As[buf][row1][c81], &A[(size_t)(m0 + row1) * K + (k0) + c81]); \
    async_cp16(&Bs[buf][row1][c81], &Bt[(size_t)(n0 + row1) * K + (k0) + c81]);\
  } while (0)

  STAGE_TILE(0, 0);

  const int nk = K >> 5;
  for (int kt = 0; kt < nk; ++kt) {
    const int cur = kt & 1;
    async_wait0();        // this wave's async copies into `cur` have landed
    __syncthreads();      // ... and everyone else's too
    if (kt + 1 < nk) STAGE_TILE(cur ^ 1, (kt + 1) << 5);  // overlap next tile

    FragB a[2], b[4];
#pragma unroll
    for (int tm = 0; tm < 2; ++tm) {
      int r = wm * 32 + tm * 16 + nn;  // A row = lane%16
      a[tm].u[0] = *(const uint4*)&As[cur][r][8 * hl];
      a[tm].u[1] = *(const uint4*)&As[cur][r][16 + 8 * hl];
    }
#pragma unroll
    for (int tn = 0; tn < 4; ++tn) {
      int r = wn * 64 + tn * 16 + nn;  // B column = lane%16
      b[tn].u[0] = *(const uint4*)&Bs[cur][r][8 * hl];
      b[tn].u[1] = *(const uint4*)&Bs[cur][r][16 + 8 * hl];
    }
#pragma unroll
    for (int tm = 0; tm < 2; ++tm)
#pragma unroll
      for (int tn = 0; tn < 4; ++tn)
        acc[tm][tn] = wmma_bf16(a[tm].v, b[tn].v, acc[tm][tn]);
  }
#undef STAGE_TILE

  // Epilogue. C layout: VGPR v -> M = v + 8*hl ; N = lane%16
#pragma unroll
  for (int tm = 0; tm < 2; ++tm) {
#pragma unroll
    for (int tn = 0; tn < 4; ++tn) {
      int gn = n0 + wn * 64 + tn * 16 + nn;
      float bv = bias[gn];
#pragma unroll
      for (int v = 0; v < 8; ++v) {
        int gm = m0 + wm * 32 + tm * 16 + v + 8 * hl;
        float val = acc[tm][tn][v] + bv;
        if (MODE == 0) {
          // n -> (three, head, dh); m -> (batch, seq)
          int three = gn >> 9, h = (gn >> 6) & 7, dh = gn & 63;
          int bb = gm >> 12, s = gm & 4095;
          int bh = bb * 8 + h;
          u16b hv = f2bf(val);
          if (three == 0)
            qo[((size_t)bh * 4096 + s) * 64 + dh] = hv;
          else if (three == 1)
            ko[((size_t)bh * 4096 + s) * 64 + dh] = hv;
          else
            vto[((size_t)bh * 64 + dh) * 4096 + s] = hv;  // V transposed
        } else {
          outf[(size_t)gm * N + gn] = val;
        }
      }
    }
  }
}

// ---------------------------------------------------------------------------
// Sliding-window attention. Block = 64 queries of one (b,h); 4 waves x 16 q.
// Wave w: queries qs=s0+16w, padded key window [qs-32, qs+63] (96 keys).
// S = Q*K^T (12 wmma) -> mask/softmax -> P(bf16 via LDS) -> O = P*V (12 wmma)
// ---------------------------------------------------------------------------
__global__ __launch_bounds__(128) void attn_kernel(const u16b* __restrict__ Q,
                                                   const u16b* __restrict__ Kc,
                                                   const u16b* __restrict__ Vt,
                                                   u16b* __restrict__ attnb) {
  extern __shared__ u16b smem[];
  u16b* Qs = smem;                //  [64][72]
  u16b* Ks = Qs + 64 * 72;        // [176][72]  keys s0-32 .. s0+143
  u16b* Vs = Ks + 176 * 72;       //  [64][184] V^T: [dh][key]
  u16b* Ps = Vs + 64 * 184;       // [4][16][104] per-wave P stripes

  const int tid = threadIdx.x, lane = tid & 31, w = tid >> 5;
  const int hl = lane >> 4, nn = lane & 15;
  const int bid = blockIdx.x;
  const int tile = bid & 63, bh = bid >> 6;
  const int s0 = tile * 64;
  const size_t qkbase = (size_t)bh * 4096 * 64;

  // ---- stage Q: 64x64 (512 chunks of 8 halfs) ----
#pragma unroll
  for (int i = 0; i < 4; ++i) {
    int c = tid + i * 128;
    int row = c >> 3, c8 = (c & 7) << 3;
    async_cp16(&Qs[row * 72 + c8], &Q[qkbase + (size_t)(s0 + row) * 64 + c8]);
  }
  // ---- stage K: 176x64, rows clamped (clamped rows are fully masked later) --
#pragma unroll
  for (int i = 0; i < 11; ++i) {
    int c = tid + i * 128;
    if (c < 1408) {
      int row = c >> 3, c8 = (c & 7) << 3;
      int s = s0 - 32 + row;
      s = s < 0 ? 0 : (s > 4095 ? 4095 : s);
      async_cp16(&Ks[row * 72 + c8], &Kc[qkbase + (size_t)s * 64 + c8]);
    }
  }
  // ---- stage V^T: 64 rows(dh) x 176 cols(keys); 8-aligned clamp ----
#pragma unroll
  for (int i = 0; i < 11; ++i) {
    int c = tid + i * 128;
    if (c < 1408) {
      int row = c / 22, cc = c - row * 22;
      int c8 = cc << 3;
      int sstart = s0 - 32 + c8;
      sstart = sstart < 0 ? 0 : (sstart > 4096 - 8 ? 4096 - 8 : sstart);
      async_cp16(&Vs[row * 184 + c8],
                 &Vt[(size_t)(bh * 64 + row) * 4096 + sstart]);
    }
  }
  async_wait0();
  __syncthreads();

  const int qs = s0 + w * 16;  // wave's first query (global)

  // Q A-fragments (dh split 0..31 / 32..63)
  FragB aq[2];
  {
    int r = w * 16 + nn;
    aq[0].u[0] = *(const uint4*)&Qs[r * 72 + 8 * hl];
    aq[0].u[1] = *(const uint4*)&Qs[r * 72 + 16 + 8 * hl];
    aq[1].u[0] = *(const uint4*)&Qs[r * 72 + 32 + 8 * hl];
    aq[1].u[1] = *(const uint4*)&Qs[r * 72 + 48 + 8 * hl];
  }

  // ---- scores: 6 key tiles x (K=64 -> 2 wmma) ----
  v8f st[6];
#pragma unroll
  for (int tn = 0; tn < 6; ++tn) {
    v8f c = zero8();
    FragB b0, b1;
    int r = (w * 16 + tn * 16 + nn) * 72;  // key row (block-relative)
    b0.u[0] = *(const uint4*)&Ks[r + 8 * hl];
    b0.u[1] = *(const uint4*)&Ks[r + 16 + 8 * hl];
    b1.u[0] = *(const uint4*)&Ks[r + 32 + 8 * hl];
    b1.u[1] = *(const uint4*)&Ks[r + 48 + 8 * hl];
    c = wmma_bf16(aq[0].v, b0.v, c);
    c = wmma_bf16(aq[1].v, b1.v, c);
    st[tn] = c;
  }

  // ---- band + boundary mask, row softmax (rows live in 16-lane groups) ----
  const float scale = 0.125f;  // 1/sqrt(64)
  float rmax[8], rsum[8];
#pragma unroll
  for (int v = 0; v < 8; ++v) rmax[v] = -3.0e38f;
#pragma unroll
  for (int tn = 0; tn < 6; ++tn) {
#pragma unroll
    for (int v = 0; v < 8; ++v) {
      int qrow = qs + v + 8 * hl;
      int kpos = qs - 32 + tn * 16 + nn;
      float sv = st[tn][v] * scale;
      bool valid = (kpos >= 0) && (kpos < 4096) && (kpos >= qrow - 32) &&
                   (kpos <= qrow + 32);
      sv = valid ? sv : -3.0e38f;
      st[tn][v] = sv;
      rmax[v] = fmaxf(rmax[v], sv);
    }
  }
#pragma unroll
  for (int v = 0; v < 8; ++v) {
    float x = rmax[v];
    x = fmaxf(x, SWZ_XOR_F(x, 1));
    x = fmaxf(x, SWZ_XOR_F(x, 2));
    x = fmaxf(x, SWZ_XOR_F(x, 4));
    x = fmaxf(x, SWZ_XOR_F(x, 8));
    rmax[v] = x;
    rsum[v] = 0.0f;
  }
#pragma unroll
  for (int tn = 0; tn < 6; ++tn) {
#pragma unroll
    for (int v = 0; v < 8; ++v) {
      float p = __expf(st[tn][v] - rmax[v]);
      st[tn][v] = p;
      rsum[v] += p;
    }
  }
#pragma unroll
  for (int v = 0; v < 8; ++v) {
    float x = rsum[v];
    x += SWZ_XOR_F(x, 1);
    x += SWZ_XOR_F(x, 2);
    x += SWZ_XOR_F(x, 4);
    x += SWZ_XOR_F(x, 8);
    rsum[v] = 1.0f / x;
  }

  // ---- write P (bf16) to per-wave LDS stripe: C-layout -> row-major ----
  u16b* Pw = Ps + w * 16 * 104;
#pragma unroll
  for (int tn = 0; tn < 6; ++tn) {
#pragma unroll
    for (int v = 0; v < 8; ++v) {
      int row = v + 8 * hl;
      Pw[row * 104 + tn * 16 + nn] = f2bf(st[tn][v] * rsum[v]);
    }
  }
  __syncthreads();

  // ---- O = P[16x96] * V[96x64] ----
  FragB ap[3];
#pragma unroll
  for (int kf = 0; kf < 3; ++kf) {
    int r = nn * 104;
    ap[kf].u[0] = *(const uint4*)&Pw[r + kf * 32 + 8 * hl];
    ap[kf].u[1] = *(const uint4*)&Pw[r + kf * 32 + 16 + 8 * hl];
  }
  const int bb = bh >> 3, h = bh & 7;
#pragma unroll
  for (int tn2 = 0; tn2 < 4; ++tn2) {
    v8f o = zero8();
#pragma unroll
    for (int kf = 0; kf < 3; ++kf) {
      FragB bv;
      int r = (tn2 * 16 + nn) * 184;          // dh row of V^T
      int cb = w * 16 + kf * 32 + 8 * hl;     // key offset in block window
      bv.u[0] = *(const uint4*)&Vs[r + cb];
      bv.u[1] = *(const uint4*)&Vs[r + cb + 16];
      o = wmma_bf16(ap[kf].v, bv.v, o);
    }
#pragma unroll
    for (int v = 0; v < 8; ++v) {
      int sG = qs + v + 8 * hl;
      int dh = tn2 * 16 + nn;
      attnb[((size_t)(bb * 4096 + sG)) * 512 + h * 64 + dh] = f2bf(o[v]);
    }
  }
}

// ---------------------------------------------------------------------------
// Launch
// ---------------------------------------------------------------------------
extern "C" void kernel_launch(void* const* d_in, const int* in_sizes, int n_in,
                              void* d_out, int out_size, void* d_ws,
                              size_t ws_size, hipStream_t stream) {
  const float* x     = (const float*)d_in[0];  // [4,4096,512]
  const float* Wqkv  = (const float*)d_in[1];  // [512,1536]
  const float* bqkv  = (const float*)d_in[2];  // [1536]
  const float* Wproj = (const float*)d_in[3];  // [512,512]
  const float* bproj = (const float*)d_in[4];  // [512]
  float* out = (float*)d_out;                  // [4,4096,512]

  char* ws = (char*)d_ws;
  size_t off = 0;
  u16b* xb     = (u16b*)(ws + off); off += (size_t)16384 * 512 * 2;  // x bf16
  u16b* wqkvt  = (u16b*)(ws + off); off += (size_t)1536 * 512 * 2;   // Wqkv^T
  u16b* wprojt = (u16b*)(ws + off); off += (size_t)512 * 512 * 2;    // Wproj^T
  u16b* q      = (u16b*)(ws + off); off += (size_t)32 * 4096 * 64 * 2;
  u16b* k      = (u16b*)(ws + off); off += (size_t)32 * 4096 * 64 * 2;
  u16b* vt     = (u16b*)(ws + off); off += (size_t)32 * 64 * 4096 * 2;
  u16b* attnb  = (u16b*)(ws + off); off += (size_t)16384 * 512 * 2;

  // 1) conversions
  cvt_f32_bf16_kernel<<<4096, 256, 0, stream>>>(x, xb, 16384 * 512);
  transpose_cvt_kernel<<<1024, 256, 0, stream>>>(Wqkv, wqkvt, 512, 1536);
  transpose_cvt_kernel<<<512, 256, 0, stream>>>(Wproj, wprojt, 512, 512);

  // 2) QKV GEMM (M=16384, N=1536, K=512) with scatter epilogue
  gemm_bf16_kernel<0><<<dim3(12, 128), 256, 0, stream>>>(
      xb, wqkvt, bqkv, q, k, vt, nullptr, 16384, 1536, 512);

  // 3) sliding-window attention (32 bh * 64 query-tiles)
  size_t shmem = (size_t)(64 * 72 + 176 * 72 + 64 * 184 + 4 * 16 * 104) * 2;
  (void)hipFuncSetAttribute((const void*)attn_kernel,
                            hipFuncAttributeMaxDynamicSharedMemorySize,
                            (int)shmem);
  attn_kernel<<<2048, 128, shmem, stream>>>(q, k, vt, attnb);

  // 4) output projection (M=16384, N=512, K=512), fp32 out + bias
  gemm_bf16_kernel<1><<<dim3(4, 128), 256, 0, stream>>>(
      attnb, wprojt, bproj, nullptr, nullptr, nullptr, out, 16384, 512, 512);
}
